// GeneralizedCoordinateHierarchicalVAE_81930796138501
// MI455X (gfx1250) — matrix operand order repaired and checked by reading, not compiled, not run on hardware
//
#include <hip/hip_runtime.h>
#include <hip/hip_fp16.h>

// Problem constants
#define B_     256
#define H_     1024
#define SEQ_   200
#define COORD_ 8

// Launch shape: persistent cooperative grid
#define NWG    64
#define TPB    256
#define NWAVES (NWG * (TPB / 32))   // 512 wave32 waves

typedef _Float16 v16h __attribute__((ext_vector_type(16)));
typedef _Float16 v8h  __attribute__((ext_vector_type(8)));
typedef float    v8f  __attribute__((ext_vector_type(8)));

#define WMMA_F16(a, b, c) \
  __builtin_amdgcn_wmma_f32_16x16x32_f16(false, (a), false, (b), (short)0, (c), false, false)

// Native-rate activations: 1 x v_exp_f32 + 1 x v_rcp_f32 each (no IEEE div sequence).
__device__ __forceinline__ float fast_sigmoid(float x) {
  return __builtin_amdgcn_rcpf(1.0f + __expf(-x));
}
__device__ __forceinline__ float fast_tanh(float x) {
  return 1.0f - 2.0f * __builtin_amdgcn_rcpf(1.0f + __expf(2.0f * x));
}

__device__ __forceinline__ v8f splat8(float v) {
  v8f r;
#pragma unroll
  for (int i = 0; i < 8; ++i) r[i] = v;
  return r;
}

// A fragment: 16x32 f16 tile of X (row-major, leading dim ld).
// Lane L: row = rowbase + (L&15); 16 contiguous K values starting at k0 + (L>>4)*16.
__device__ __forceinline__ v16h frag_a(const _Float16* __restrict__ X, int ld,
                                       int rowbase, int k0, int lane) {
  const int row = rowbase + (lane & 15);
  const _Float16* p = X + (size_t)row * ld + k0 + ((lane >> 4) << 4);
  v16h a;
  ((v8h*)&a)[0] = *(const v8h*)p;
  ((v8h*)&a)[1] = *(const v8h*)(p + 8);
  return a;
}

// B fragment: 32x16 f16, B[k][n] = W[colbase+n][k] with W row-major (N,K), leading dim ld.
// Lane L: col = colbase + (L&15); 16 contiguous K values starting at k0 + (L>>4)*16.
__device__ __forceinline__ v16h frag_b(const _Float16* __restrict__ W, int ld,
                                       int colbase, int k0, int lane) {
  const int col = colbase + (lane & 15);
  return *(const v16h*)(W + (size_t)col * ld + k0 + ((lane >> 4) << 4));
}

// gfx1250 async copy: stage a contiguous tile into LDS on the ASYNCcnt path.
template <int NBYTES>
__device__ __forceinline__ void async_copy(unsigned lds_base,
                                           const _Float16* __restrict__ g, int tid) {
#pragma unroll
  for (int i = 0; i < NBYTES / (TPB * 16); ++i) {
    const unsigned byte_off = (unsigned)(tid + i * TPB) * 16u;
    const unsigned ldsa = lds_base + byte_off;
    const unsigned long long ga = (unsigned long long)(const char*)g + byte_off;
    asm volatile("global_load_async_to_lds_b128 %0, %1, off"
                 :: "v"(ldsa), "v"(ga)
                 : "memory");
  }
}

__device__ __forceinline__ void async_wait_all() {
  asm volatile("s_wait_asynccnt 0x0" ::: "memory");
}

// One 32x16 output unit against a 4-quadrant gate matrix W (N=4H), K = H.
// A tile (32 rows) comes from LDS; B fragments are global, ping-pong double-buffered
// (K unrolled by 2) so the rotation costs zero register moves / hazard NOPs.
__device__ __forceinline__ void gemm_band4(const _Float16* __restrict__ sA,
                                           const _Float16* __restrict__ W,
                                           int nb, int lane, v8f acc[4][2]) {
  v16h bA[4], bB[4];
#pragma unroll
  for (int q = 0; q < 4; ++q) bA[q] = frag_b(W, H_, q * H_ + nb, 0, lane);
#pragma unroll 1
  for (int kc = 0; kc < H_ / 32; kc += 2) {
    const int k0 = kc << 5;
    // even half: prefetch bB for k0+32, consume bA
#pragma unroll
    for (int q = 0; q < 4; ++q) bB[q] = frag_b(W, H_, q * H_ + nb, k0 + 32, lane);
    {
      const v16h a0 = frag_a(sA, H_, 0, k0, lane);
      const v16h a1 = frag_a(sA, H_, 16, k0, lane);
#pragma unroll
      for (int q = 0; q < 4; ++q) {
        acc[q][0] = WMMA_F16(a0, bA[q], acc[q][0]);
        acc[q][1] = WMMA_F16(a1, bA[q], acc[q][1]);
      }
    }
    // odd half: prefetch bA for k0+64 (wrap-safe dead load on last iter), consume bB
    const int k2 = (kc + 2 < H_ / 32) ? (k0 + 64) : 0;
#pragma unroll
    for (int q = 0; q < 4; ++q) bA[q] = frag_b(W, H_, q * H_ + nb, k2, lane);
    {
      const v16h a0 = frag_a(sA, H_, 0, k0 + 32, lane);
      const v16h a1 = frag_a(sA, H_, 16, k0 + 32, lane);
#pragma unroll
      for (int q = 0; q < 4; ++q) {
        acc[q][0] = WMMA_F16(a0, bB[q], acc[q][0]);
        acc[q][1] = WMMA_F16(a1, bB[q], acc[q][1]);
      }
    }
  }
}

// Sense-reversing grid barrier: bar[0] = arrival count, bar[1] = generation.
__device__ void grid_sync(unsigned* bar) {
  __threadfence();
  __syncthreads();
  if (threadIdx.x == 0) {
    unsigned gen  = __hip_atomic_load(bar + 1, __ATOMIC_RELAXED, __HIP_MEMORY_SCOPE_AGENT);
    unsigned prev = __hip_atomic_fetch_add(bar, 1u, __ATOMIC_ACQ_REL, __HIP_MEMORY_SCOPE_AGENT);
    if (prev == NWG - 1) {
      __hip_atomic_store(bar, 0u, __ATOMIC_RELAXED, __HIP_MEMORY_SCOPE_AGENT);
      __hip_atomic_fetch_add(bar + 1, 1u, __ATOMIC_RELEASE, __HIP_MEMORY_SCOPE_AGENT);
    } else {
      while (__hip_atomic_load(bar + 1, __ATOMIC_ACQUIRE, __HIP_MEMORY_SCOPE_AGENT) == gen) {
        __builtin_amdgcn_s_sleep(2);
      }
    }
  }
  __syncthreads();
  __threadfence();
}

__global__ __launch_bounds__(TPB) void vae_lstm_kernel(
    const float* __restrict__ zp,   const float* __restrict__ zs,   const float* __restrict__ zst,
    const float* __restrict__ Wproj,const float* __restrict__ bproj,
    const float* __restrict__ Wih0, const float* __restrict__ Whh0,
    const float* __restrict__ bih0, const float* __restrict__ bhh0,
    const float* __restrict__ Wih1, const float* __restrict__ Whh1,
    const float* __restrict__ bih1, const float* __restrict__ bhh1,
    const float* __restrict__ Wo1,  const float* __restrict__ bo1,
    const float* __restrict__ Wo2,  const float* __restrict__ bo2,
    float* __restrict__ out,
    unsigned* bar,
    _Float16* __restrict__ Whh0h, _Float16* __restrict__ Wih1h, _Float16* __restrict__ Whh1h,
    _Float16* __restrict__ Wo1h,  _Float16* __restrict__ Wih0h, _Float16* __restrict__ Wo2h,
    float* __restrict__ c1, float* __restrict__ c2,
    _Float16* __restrict__ h1h, _Float16* __restrict__ h2h,
    float* __restrict__ xbuf) {

  const int tid  = threadIdx.x;
  const int lane = tid & 31;
  const int wave = tid >> 5;
  const int wg   = blockIdx.x;
  const int kh   = lane >> 4;                    // lane half (C/D row group)
  const int colc = lane & 15;                    // C/D column within tile
  const int gt   = wg * TPB + tid;               // global thread id
  const int GS   = NWG * TPB;                    // grid-stride

  // WG-cooperative ownership: one 32-row M-pair + 8 n-tiles per WG (1 per wave).
  // 8 M-pairs x 8 WGs = 64 WGs; 512 waves x (32x16 unit) = 256 x 4096 output.
  const int mp      = wg >> 3;                   // M-pair 0..7 (rows mp*32 .. +31)
  const int rowbase = mp << 5;
  const int nb      = ((wg & 7) * 8 + wave) << 4;  // n-tile column base (within H)

  // 64 KB LDS: phase1/2 = 32-row A tile; phase3 = 16-row A tile + relu staging
  __shared__ __align__(128) _Float16 sbuf[32768];
  const _Float16* sA   = sbuf;
  _Float16*       rlds = sbuf + 16384;
  const unsigned  ldsb = (unsigned)(size_t)(void*)sbuf;

  // ---------------- Phase -1: convert weights to f16 (L2-resident working set) ---------
  for (int i = gt; i < 4096 * 1024; i += GS) Whh0h[i] = (_Float16)Whh0[i];
  for (int i = gt; i < 4096 * 1024; i += GS) Wih1h[i] = (_Float16)Wih1[i];
  for (int i = gt; i < 4096 * 1024; i += GS) Whh1h[i] = (_Float16)Whh1[i];
  for (int i = gt; i <  512 * 1024; i += GS) Wo1h[i]  = (_Float16)Wo1[i];
  for (int i = gt; i < 4096 * 8;    i += GS) Wih0h[i] = (_Float16)Wih0[i];
  for (int i = gt; i <    8 * 512;  i += GS) Wo2h[i]  = (_Float16)Wo2[i];

  // ---------------- Phase 0: init = z @ Wproj^T + bproj (K=224, VALU) ------------------
  for (int j = gt; j < B_ * 2 * H_; j += GS) {
    const int b   = j >> 11;
    const int col = j & 2047;
    const float* w = Wproj + (size_t)col * 224;
    float acc = bproj[col];
    const float* z0 = zp  + b * 32;
    const float* z1 = zs  + b * 64;
    const float* z2 = zst + b * 128;
    for (int k = 0; k < 32;  ++k) acc += z0[k] * w[k];
    for (int k = 0; k < 64;  ++k) acc += z1[k] * w[32 + k];
    for (int k = 0; k < 128; ++k) acc += z2[k] * w[96 + k];
    if (col < H_) {
      const int idx = b * H_ + col;
      h1h[idx] = (_Float16)acc;                  // parity-0 buffer
      h2h[idx] = (_Float16)acc;
    } else {
      const int idx = b * H_ + (col - H_);
      c1[idx] = acc;
      c2[idx] = acc;
    }
  }
  for (int j = gt; j < B_ * COORD_; j += GS) xbuf[j] = 0.0f;
  grid_sync(bar);

  // ---------------- Time loop: 200 serialized steps, 3 grid barriers/step --------------
  for (int t = 0; t < SEQ_; ++t) {
    const int p = t & 1;
    const _Float16* h1r = h1h + (size_t)p       * (B_ * H_);
    _Float16*       h1w = h1h + (size_t)(1 - p) * (B_ * H_);
    const _Float16* h2r = h2h + (size_t)p       * (B_ * H_);
    _Float16*       h2w = h2h + (size_t)(1 - p) * (B_ * H_);

    // ===== Phase 1: LSTM layer 0 =====================================================
    async_copy<65536>(ldsb, h1r + (size_t)rowbase * H_, tid);   // 32-row A tile
    async_wait_all();
    __syncthreads();
    {
      v8f acc[4][2];
      // accumulator init: bias + tiny x @ Wih0^T (K = 8, VALU)
#pragma unroll
      for (int q = 0; q < 4; ++q) {
        const int gc = q * H_ + nb + colc;
        const float bv = bih0[gc] + bhh0[gc];
        const v8h wv = *(const v8h*)(Wih0h + (size_t)gc * COORD_);
#pragma unroll
        for (int mt = 0; mt < 2; ++mt) {
#pragma unroll
          for (int r = 0; r < 8; ++r) {
            const float* xr = xbuf + (rowbase + (mt << 4) + (kh << 3) + r) * COORD_;
            float s = bv;
#pragma unroll
            for (int k = 0; k < 8; ++k) s += xr[k] * (float)wv[k];
            acc[q][mt][r] = s;
          }
        }
      }
      gemm_band4(sA, Whh0h, nb, lane, acc);
      // fused LSTM cell update (i,f,g,o)
#pragma unroll
      for (int mt = 0; mt < 2; ++mt) {
#pragma unroll
        for (int r = 0; r < 8; ++r) {
          const int row = rowbase + (mt << 4) + (kh << 3) + r;
          const int idx = row * H_ + nb + colc;
          const float ig = fast_sigmoid(acc[0][mt][r]);
          const float fg = fast_sigmoid(acc[1][mt][r]);
          const float gg = fast_tanh(acc[2][mt][r]);
          const float og = fast_sigmoid(acc[3][mt][r]);
          const float cn = fg * c1[idx] + ig * gg;
          c1[idx]  = cn;
          h1w[idx] = (_Float16)(og * fast_tanh(cn));
        }
      }
    }
    grid_sync(bar);

    // ===== Phase 2: LSTM layer 1: two GEMMs sequentially against one LDS A tile ======
    {
      v8f acc[4][2];
#pragma unroll
      for (int q = 0; q < 4; ++q) {
        const int gc = q * H_ + nb + colc;
        const float bv = bih1[gc] + bhh1[gc];
        acc[q][0] = splat8(bv);
        acc[q][1] = splat8(bv);
      }
      async_copy<65536>(ldsb, h1w + (size_t)rowbase * H_, tid);  // h1_new tile
      async_wait_all();
      __syncthreads();
      gemm_band4(sA, Wih1h, nb, lane, acc);
      __syncthreads();                                           // all LDS reads done
      async_copy<65536>(ldsb, h2r + (size_t)rowbase * H_, tid);  // h2_old tile
      async_wait_all();
      __syncthreads();
      gemm_band4(sA, Whh1h, nb, lane, acc);
#pragma unroll
      for (int mt = 0; mt < 2; ++mt) {
#pragma unroll
        for (int r = 0; r < 8; ++r) {
          const int row = rowbase + (mt << 4) + (kh << 3) + r;
          const int idx = row * H_ + nb + colc;
          const float ig = fast_sigmoid(acc[0][mt][r]);
          const float fg = fast_sigmoid(acc[1][mt][r]);
          const float gg = fast_tanh(acc[2][mt][r]);
          const float og = fast_sigmoid(acc[3][mt][r]);
          const float cn = fg * c2[idx] + ig * gg;
          c2[idx]  = cn;
          h2w[idx] = (_Float16)(og * fast_tanh(cn));
        }
      }
    }
    grid_sync(bar);

    // ===== Phase 3: output head, owned per 16-row M-tile by WGs 0..15 ================
    if (wg < 16) {
      const int rb = wg << 4;
      async_copy<32768>(ldsb, h2w + (size_t)rb * H_, tid);   // 16-row A tile: h2_new
      async_wait_all();
      __syncthreads();
      // r = relu(h2_new @ Wo1^T + bo1): each wave covers 4 N-tiles (64 cols), WG = 512
      for (int j = 0; j < 4; ++j) {
        const int nbh = ((wave << 2) + j) << 4;
        v8f acc = splat8(bo1[nbh + colc]);
        v16h bA = frag_b(Wo1h, H_, nbh, 0, lane);
#pragma unroll 1
        for (int kc = 0; kc < H_ / 32; kc += 2) {
          const int k0 = kc << 5;
          const v16h bB = frag_b(Wo1h, H_, nbh, k0 + 32, lane);
          acc = WMMA_F16(frag_a(sA, H_, 0, k0, lane), bA, acc);
          const int k2 = (kc + 2 < H_ / 32) ? (k0 + 64) : 0;
          bA = frag_b(Wo1h, H_, nbh, k2, lane);
          acc = WMMA_F16(frag_a(sA, H_, 0, k0 + 32, lane), bB, acc);
        }
#pragma unroll
        for (int r = 0; r < 8; ++r) {
          const float v = acc[r];
          rlds[(r + (kh << 3)) * 512 + nbh + colc] = (_Float16)(v > 0.0f ? v : 0.0f);
        }
      }
      __syncthreads();
      // out = r @ Wo2^T + bo2: M=16, K=512, N=16 (8 valid, zero-padded B)
      if (wave == 0) {
        v8f acc = splat8(0.0f);
        for (int kc = 0; kc < 512 / 32; ++kc) {
          const int k0 = kc << 5;
          const v16h a = frag_a(rlds, 512, 0, k0, lane);
          v16h bb = {};
          if (colc < 8) bb = *(const v16h*)(Wo2h + (size_t)colc * 512 + k0 + (kh << 4));
          acc = WMMA_F16(a, bb, acc);
        }
#pragma unroll
        for (int r = 0; r < 8; ++r) {
          const int b = rb + r + (kh << 3);
          if (colc < 8) {
            const float v = acc[r] + bo2[colc];
            out[((size_t)b * SEQ_ + t) * COORD_ + colc] = v;  // (B, SEQ, COORD)
            xbuf[b * COORD_ + colc] = v;                      // autoregressive feedback
          }
        }
      }
      __syncthreads();
    }
    grid_sync(bar);  // x_{t+1}, ready for next step's Phase 1
  }
}

extern "C" void kernel_launch(void* const* d_in, const int* in_sizes, int n_in,
                              void* d_out, int out_size, void* d_ws, size_t ws_size,
                              hipStream_t stream) {
  (void)in_sizes; (void)n_in; (void)out_size; (void)ws_size;
  const float* zp    = (const float*)d_in[0];
  const float* zs    = (const float*)d_in[1];
  const float* zst   = (const float*)d_in[2];
  const float* Wproj = (const float*)d_in[3];
  const float* bproj = (const float*)d_in[4];
  const float* Wih0  = (const float*)d_in[5];
  const float* Whh0  = (const float*)d_in[6];
  const float* bih0  = (const float*)d_in[7];
  const float* bhh0  = (const float*)d_in[8];
  const float* Wih1  = (const float*)d_in[9];
  const float* Whh1  = (const float*)d_in[10];
  const float* bih1  = (const float*)d_in[11];
  const float* bhh1  = (const float*)d_in[12];
  const float* Wo1   = (const float*)d_in[13];
  const float* bo1   = (const float*)d_in[14];
  const float* Wo2   = (const float*)d_in[15];
  const float* bo2   = (const float*)d_in[16];
  float* out = (float*)d_out;

  // Workspace layout (≈30 MB): barrier | f16 weights | c1 c2 | h1 h2 ping-pong | x
  char* ws = (char*)d_ws;
  size_t off = 256;
  auto take = [&](size_t bytes) -> char* {
    char* p = ws + off;
    off += (bytes + 255) & ~(size_t)255;
    return p;
  };
  unsigned* bar   = (unsigned*)ws;
  _Float16* Whh0h = (_Float16*)take((size_t)4096 * 1024 * 2);
  _Float16* Wih1h = (_Float16*)take((size_t)4096 * 1024 * 2);
  _Float16* Whh1h = (_Float16*)take((size_t)4096 * 1024 * 2);
  _Float16* Wo1h  = (_Float16*)take((size_t)512 * 1024 * 2);
  _Float16* Wih0h = (_Float16*)take((size_t)4096 * 8 * 2);
  _Float16* Wo2h  = (_Float16*)take((size_t)8 * 512 * 2);
  float*    c1    = (float*)take((size_t)B_ * H_ * 4);
  float*    c2    = (float*)take((size_t)B_ * H_ * 4);
  _Float16* h1h   = (_Float16*)take((size_t)2 * B_ * H_ * 2);
  _Float16* h2h   = (_Float16*)take((size_t)2 * B_ * H_ * 2);
  float*    xbuf  = (float*)take((size_t)B_ * COORD_ * 4);

  hipMemsetAsync(d_ws, 0, 256, stream);  // barrier count/generation (ws is poisoned)

  vae_lstm_kernel<<<NWG, TPB, 0, stream>>>(
      zp, zs, zst, Wproj, bproj, Wih0, Whh0, bih0, bhh0,
      Wih1, Whh1, bih1, bhh1, Wo1, bo1, Wo2, bo2, out,
      bar, Whh0h, Wih1h, Whh1h, Wo1h, Wih0h, Wo2h,
      c1, c2, h1h, h2h, xbuf);
}
